// Model_64587718197886
// MI455X (gfx1250) — matrix operand order repaired and checked by reading, not compiled, not run on hardware
//
#include <hip/hip_runtime.h>
#include <hip/hip_bf16.h>
#include <math.h>

typedef __attribute__((ext_vector_type(16))) _Float16 v16h;
typedef __attribute__((ext_vector_type(8)))  float    v8f;

#define N_NODES 140
#define TT 12
#define BB 32
#define HID 140
#define HEADS 5
#define HD 28
#define HORIZON 3
#define CONV_HID 32
#define GP 144          // per-gate padded width
#define GT (3*GP)       // 432
#define KX 128          // x feature K
#define KXP 136         // padded pitch for K=128 arrays (bank-conflict-free, 16B-aligned)
#define KHV 160         // padded hidden K (140 -> 5 k-steps of 32)
#define KHP 168         // padded pitch for K=160 arrays
#define AK 288          // attention K (280 -> 9 k-steps)
#define AKP 296         // padded pitch for K=288 arrays
#define ROWS_BT (BB*TT) // 384

// ---- CDNA5 WMMA fragment loaders (wave32, v_wmma_f32_16x16x32_f16) ----
// A (16x32 f16), row-major LDS [rows][pitch]: lane m=l&15 = row, kh=l>>4.
// Element v<4 -> K = 2v+p + 8kh ; v>=4 -> 16+2(v-4)+p + 8kh.
// Both 8-half groups are contiguous -> two ds_load_b128 per fragment.
__device__ __forceinline__ v16h lds_load_a(const _Float16* s, int row_base, int pitch, int k_base) {
  int l = threadIdx.x & 31;
  int m = l & 15, kh = l >> 4;
  const _Float16* p = s + (size_t)(row_base + m) * pitch + k_base + 8 * kh;
  v16h a;
#pragma unroll
  for (int e = 0; e < 16; ++e) {
    int v = e >> 1, pb = e & 1;
    int k = (v < 4) ? (2 * v + pb) : (16 + 2 * (v - 4) + pb);
    a[e] = p[k];
  }
  return a;
}
// B (32x16 f16) from COLUMN-MAJOR LDS [cols][pitch]: lane c=l&15 = column,
// K = 8*(l>>4) + (e<8 ? e : e+8). Two 8-half groups -> two ds_load_b128, no packing movs.
__device__ __forceinline__ v16h lds_load_bT(const _Float16* s, int col_base, int pitch, int k_base) {
  int l = threadIdx.x & 31;
  int c = l & 15, kh = l >> 4;
  const _Float16* p = s + (size_t)(col_base + c) * pitch + k_base + 8 * kh;
  v16h b;
#pragma unroll
  for (int e = 0; e < 16; ++e) b[e] = p[(e < 8) ? e : (e + 8)];
  return b;
}
#define WMMA_F16(a, b, acc) \
  __builtin_amdgcn_wmma_f32_16x16x32_f16(false, (a), false, (b), (short)0, (acc), false, false)

// ============ K1: x_sup = relu(relu(x*w1+b1) @ w2^T + b2), stored f16 ============
__global__ void __launch_bounds__(256) k_seq1(
    const float* __restrict__ x, const float* __restrict__ w1,
    const float* __restrict__ b1, const float* __restrict__ w2,
    const float* __restrict__ b2, _Float16* __restrict__ x_sup)
{
  __shared__ _Float16 h1[16 * 72];    // [16 rows][K=64, pitch 72]
  __shared__ _Float16 w2t[128 * 72];  // column-major [128 cols][K=64, pitch 72]
  int tid = threadIdx.x;
  for (int i = tid; i < 128 * 64; i += 256) {
    int h = i >> 6, e = i & 63;
    w2t[h * 72 + e] = (_Float16)w2[h * 64 + e];
  }
  int row0 = blockIdx.x * 16;
  for (int i = tid; i < 16 * 64; i += 256) {
    int r = i >> 6, e = i & 63;
    int gr = row0 + r;
    int n = gr / (BB * TT); int rem = gr % (BB * TT);
    int b = rem / TT; int t = rem % TT;
    float v = x[(b * TT + t) * N_NODES + n];
    float hv = v * w1[e] + b1[e];
    h1[r * 72 + e] = (_Float16)(hv > 0.f ? hv : 0.f);
  }
  __syncthreads();
  int w = tid >> 5, lane = tid & 31;
  int c0 = w * 16;
  v8f acc = {};
#pragma unroll
  for (int ks = 0; ks < 2; ++ks) {
    v16h a  = lds_load_a(h1, 0, 72, ks * 32);
    v16h bm = lds_load_bT(w2t, c0, 72, ks * 32);
    acc = WMMA_F16(a, bm, acc);
  }
  int col = c0 + (lane & 15);
#pragma unroll
  for (int r = 0; r < 8; ++r) {
    int row = r + 8 * (lane >> 4);
    float v = acc[r] + b2[col];
    v = v > 0.f ? v : 0.f;
    x_sup[(size_t)(row0 + row) * KX + col] = (_Float16)v;
  }
}

// ============ K2: 140 per-node GRU scans, weights resident in LDS ============
__global__ void __launch_bounds__(256, 1) k_gru(
    const _Float16* __restrict__ x_sup, const float* __restrict__ wih,
    const float* __restrict__ whh, const float* __restrict__ bih,
    const float* __restrict__ bhh, float* __restrict__ outs)
{
  extern __shared__ char smem[];
  _Float16* wihT = (_Float16*)smem;             // col-major [432][136]  117.5 KB
  _Float16* whhT = wihT + (size_t)GT * KXP;     // col-major [432][168]  145.2 KB
  _Float16* h16  = whhT + (size_t)GT * KHP;     // row-major [32][168]
  _Float16* x16  = h16 + 32 * KHP;              // row-major [32][136]
  _Float16* r16  = x16 + 32 * KXP;              // [32][144]
  _Float16* z16  = r16 + 32 * GP;               // [32][144]
  float*    h32  = (float*)(z16 + 32 * GP);     // [32][144]
  int n = blockIdx.x, tid = threadIdx.x;
  for (int i = tid; i < GT * KX; i += 256) {
    int c = i / KX, k = i % KX;
    int g = c / GP, j = c % GP;
    float v = (j < HID) ? wih[((size_t)n * 420 + g * HID + j) * KX + k] : 0.f;
    wihT[(size_t)c * KXP + k] = (_Float16)v;
  }
  for (int i = tid; i < GT * KHV; i += 256) {
    int c = i / KHV, k = i % KHV;
    int g = c / GP, j = c % GP;
    float v = (j < HID && k < HID) ? whh[((size_t)n * 420 + g * HID + j) * HID + k] : 0.f;
    whhT[(size_t)c * KHP + k] = (_Float16)v;
  }
  for (int i = tid; i < 32 * KHP; i += 256) h16[i] = (_Float16)0.f;
  for (int i = tid; i < 32 * GP; i += 256) h32[i] = 0.f;
  __syncthreads();
  int w = tid >> 5, lane = tid & 31;
  for (int t = 0; t < TT; ++t) {
    for (int i = tid; i < 32 * KX; i += 256) {
      int b = i >> 7, k = i & 127;
      x16[b * KXP + k] = x_sup[(((size_t)n * BB + b) * TT + t) * KX + k];
    }
    __syncthreads();
    // --- r,z gates: fused input+recurrent GEMM, 2 row tiles x 18 col tiles ---
    for (int tile = w; tile < 36; tile += 8) {
      int mt = tile / 18, ct = tile % 18;
      int r0 = mt * 16, c0 = ct * 16;
      v8f acc = {};
#pragma unroll
      for (int ks = 0; ks < 4; ++ks) {
        v16h a  = lds_load_a(x16, r0, KXP, ks * 32);
        v16h bm = lds_load_bT(wihT, c0, KXP, ks * 32);
        acc = WMMA_F16(a, bm, acc);
      }
#pragma unroll
      for (int ks = 0; ks < 5; ++ks) {
        v16h a  = lds_load_a(h16, r0, KHP, ks * 32);
        v16h bm = lds_load_bT(whhT, c0, KHP, ks * 32);
        acc = WMMA_F16(a, bm, acc);
      }
      int c = c0 + (lane & 15);
      int g = c / GP, j = c % GP;
      float bi = (j < HID) ? bih[n * 420 + g * HID + j] : 0.f;
      float bh = (j < HID) ? bhh[n * 420 + g * HID + j] : 0.f;
#pragma unroll
      for (int r = 0; r < 8; ++r) {
        int row = r0 + r + 8 * (lane >> 4);
        float s = 1.f / (1.f + __expf(-(acc[r] + bi + bh)));
        if (g == 0) r16[row * GP + j] = (_Float16)s;
        else        z16[row * GP + j] = (_Float16)s;
      }
    }
    __syncthreads();
    // --- n gate: separate i_n / h_n accumulators (tanh(i_n + r*h_n)) ---
    v8f acci[3], acch[3];
#pragma unroll
    for (int q = 0; q < 3; ++q) {
      int tile = w + q * 8;
      if (tile < 18) {
        int mt = tile / 9, ct = tile % 9;
        int r0 = mt * 16, c0 = 2 * GP + ct * 16;
        v8f ai = {};
#pragma unroll
        for (int ks = 0; ks < 4; ++ks) {
          v16h a  = lds_load_a(x16, r0, KXP, ks * 32);
          v16h bm = lds_load_bT(wihT, c0, KXP, ks * 32);
          ai = WMMA_F16(a, bm, ai);
        }
        v8f ah = {};
#pragma unroll
        for (int ks = 0; ks < 5; ++ks) {
          v16h a  = lds_load_a(h16, r0, KHP, ks * 32);
          v16h bm = lds_load_bT(whhT, c0, KHP, ks * 32);
          ah = WMMA_F16(a, bm, ah);
        }
        acci[q] = ai; acch[q] = ah;
      }
    }
    __syncthreads();  // all gh reads of h16 done before we overwrite h
#pragma unroll
    for (int q = 0; q < 3; ++q) {
      int tile = w + q * 8;
      if (tile < 18) {
        int mt = tile / 9, ct = tile % 9;
        int r0 = mt * 16;
        int j = ct * 16 + (lane & 15);
        float bi = (j < HID) ? bih[n * 420 + 2 * HID + j] : 0.f;
        float bh = (j < HID) ? bhh[n * 420 + 2 * HID + j] : 0.f;
#pragma unroll
        for (int r = 0; r < 8; ++r) {
          int row = r0 + r + 8 * (lane >> 4);
          float inn = acci[q][r] + bi;
          float hn  = acch[q][r] + bh;
          float rg = (float)r16[row * GP + j];
          float zg = (float)z16[row * GP + j];
          float nn = tanhf(inn + rg * hn);
          float hold = h32[row * GP + j];
          float hnew = (1.f - zg) * nn + zg * hold;
          h32[row * GP + j] = hnew;
          h16[row * KHP + j] = (_Float16)hnew;
          if (j < HID)
            outs[(((size_t)n * BB + row) * TT + t) * HID + j] = hnew;
        }
      }
    }
    __syncthreads();
  }
}

// ============ K3: attention scores = v . tanh(W [hid;out_t] + b) ============
__global__ void __launch_bounds__(256, 1) k_attn_scores(
    const float* __restrict__ outs, const float* __restrict__ attn_w,
    const float* __restrict__ attn_b, const float* __restrict__ attn_v,
    float* __restrict__ scores)
{
  extern __shared__ char smem[];
  _Float16* wT = (_Float16*)smem;                    // col-major [144][296]
  _Float16* A  = wT + (size_t)GP * AKP;              // row-major [16][296]
  float* sc    = (float*)(A + 16 * AKP);             // [384]
  int n = blockIdx.x, tid = threadIdx.x;
  for (int i = tid; i < GP * AK; i += 256) {
    int h = i / AK, f = i % AK;
    wT[(size_t)h * AKP + f] =
        (_Float16)((f < 2 * HID && h < HID) ? attn_w[h * (2 * HID) + f] : 0.f);
  }
  for (int i = tid; i < ROWS_BT; i += 256) sc[i] = 0.f;
  __syncthreads();
  int w = tid >> 5, lane = tid & 31;
  for (int mt = 0; mt < 24; ++mt) {
    for (int i = tid; i < 16 * AK; i += 256) {
      int r = i / AK, f = i % AK;
      int gr = mt * 16 + r;
      int b = gr / TT, t = gr % TT;
      float v;
      if (f < HID)            v = outs[(((size_t)n * BB + b) * TT + (TT - 1)) * HID + f];
      else if (f < 2 * HID)   v = outs[(((size_t)n * BB + b) * TT + t) * HID + (f - HID)];
      else                    v = 0.f;
      A[r * AKP + f] = (_Float16)v;
    }
    __syncthreads();
    for (int ct = w; ct < 9; ct += 8) {
      int c0 = ct * 16;
      v8f acc = {};
#pragma unroll
      for (int ks = 0; ks < 9; ++ks) {
        v16h a  = lds_load_a(A, 0, AKP, ks * 32);
        v16h bm = lds_load_bT(wT, c0, AKP, ks * 32);
        acc = WMMA_F16(a, bm, acc);
      }
      int col = c0 + (lane & 15);
      if (col < HID) {
        float vc = attn_v[col], bb_ = attn_b[col];
#pragma unroll
        for (int r = 0; r < 8; ++r) {
          int row = r + 8 * (lane >> 4);
          atomicAdd(&sc[mt * 16 + row], vc * tanhf(acc[r] + bb_));
        }
      }
    }
    __syncthreads();
  }
  for (int i = tid; i < ROWS_BT; i += 256)
    scores[(size_t)n * ROWS_BT + i] = sc[i];
}

// ============ K4: softmax over T, weighted sum, +hid, LayerNorm ============
__global__ void k_weighted_ln(const float* __restrict__ outs, const float* __restrict__ scores,
                              const float* __restrict__ ln_g, const float* __restrict__ ln_b,
                              float* __restrict__ wres)
{
  int n = blockIdx.x, b = blockIdx.y, tid = threadIdx.x;
  __shared__ float pre[HID];
  __shared__ float stat[2];
  float wv[TT];
  {
    const float* s = scores + (size_t)n * ROWS_BT + b * TT;
    float mx = -1e30f;
    for (int t = 0; t < TT; ++t) mx = fmaxf(mx, s[t]);
    float sum = 0.f;
    for (int t = 0; t < TT; ++t) { wv[t] = __expf(s[t] - mx); sum += wv[t]; }
    float inv = 1.f / sum;
    for (int t = 0; t < TT; ++t) wv[t] *= inv;
  }
  if (tid < HID) {
    float acc = 0.f;
    for (int t = 0; t < TT; ++t)
      acc += wv[t] * outs[(((size_t)n * BB + b) * TT + t) * HID + tid];
    float hid = outs[(((size_t)n * BB + b) * TT + (TT - 1)) * HID + tid];
    pre[tid] = acc + hid;
  }
  __syncthreads();
  if (tid == 0) {
    float mu = 0.f;
    for (int i = 0; i < HID; ++i) mu += pre[i];
    mu /= HID;
    float var = 0.f;
    for (int i = 0; i < HID; ++i) { float d = pre[i] - mu; var += d * d; }
    var /= HID;
    stat[0] = mu; stat[1] = rsqrtf(var + 1e-5f);
  }
  __syncthreads();
  if (tid < HID)
    wres[((size_t)b * N_NODES + n) * HID + tid] =
        (pre[tid] - stat[0]) * stat[1] * ln_g[tid] + ln_b[tid];
}

// ============ K5: Q/K projections via WMMA ============
__global__ void __launch_bounds__(256, 1) k_qk(
    const float* __restrict__ wres, const float* __restrict__ wq,
    const float* __restrict__ wk, const float* __restrict__ bq,
    const float* __restrict__ bk, float* __restrict__ qd, float* __restrict__ kd)
{
  extern __shared__ char smem[];
  _Float16* wqT = (_Float16*)smem;          // col-major [144][168]
  _Float16* wkT = wqT + (size_t)GP * KHP;   // col-major [144][168]
  _Float16* A   = wkT + (size_t)GP * KHP;   // row-major [16][168]
  int mt = blockIdx.x, b = blockIdx.y, tid = threadIdx.x;
  for (int i = tid; i < GP * KHV; i += 256) {
    int h = i / KHV, kk = i % KHV;
    bool ok = (kk < HID && h < HID);
    wqT[(size_t)h * KHP + kk] = (_Float16)(ok ? wq[h * HID + kk] : 0.f);
    wkT[(size_t)h * KHP + kk] = (_Float16)(ok ? wk[h * HID + kk] : 0.f);
  }
  for (int i = tid; i < 16 * KHV; i += 256) {
    int r = i / KHV, kk = i % KHV;
    int row = mt * 16 + r;
    A[r * KHP + kk] = (_Float16)((row < N_NODES && kk < HID)
                          ? wres[((size_t)b * N_NODES + row) * HID + kk] : 0.f);
  }
  __syncthreads();
  int w = tid >> 5, lane = tid & 31;
  for (int ct = w; ct < 9; ct += 8) {
    int c0 = ct * 16;
    v8f aq = {}, ak = {};
#pragma unroll
    for (int ks = 0; ks < 5; ++ks) {
      v16h a   = lds_load_a(A, 0, KHP, ks * 32);
      v16h bq_ = lds_load_bT(wqT, c0, KHP, ks * 32);
      aq = WMMA_F16(a, bq_, aq);
      v16h bk_ = lds_load_bT(wkT, c0, KHP, ks * 32);
      ak = WMMA_F16(a, bk_, ak);
    }
    int h = c0 + (lane & 15);
    if (h < HID) {
#pragma unroll
      for (int r = 0; r < 8; ++r) {
        int row = mt * 16 + r + 8 * (lane >> 4);
        if (row < N_NODES) {
          qd[((size_t)b * N_NODES + row) * HID + h] = aq[r] + bq[h];
          kd[((size_t)b * N_NODES + row) * HID + h] = ak[r] + bk[h];
        }
      }
    }
  }
}

__global__ void k_zero(float* p, int n) {
  int i = blockIdx.x * blockDim.x + threadIdx.x;
  if (i < n) p[i] = 0.f;
}

// ============ K6: per-(b,head) attention softmax, mean-accumulated ============
__global__ void k_mha_att(const float* __restrict__ qd, const float* __restrict__ kd,
                          float* __restrict__ att)
{
  int head = blockIdx.x, b = blockIdx.y, tid = threadIdx.x;
  __shared__ float srow[N_NODES];
  __shared__ float red[2];
  const float scale = 0.18898223650461363f;  // 1/sqrt(28)
  for (int qi = 0; qi < N_NODES; ++qi) {
    if (tid < N_NODES) {
      const float* qp = qd + ((size_t)b * N_NODES + qi) * HID + head * HD;
      const float* kp = kd + ((size_t)b * N_NODES + tid) * HID + head * HD;
      float s = 0.f;
      for (int d = 0; d < HD; ++d) s += qp[d] * kp[d];
      srow[tid] = s * scale;
    }
    __syncthreads();
    if (tid == 0) {
      float mx = -1e30f;
      for (int i = 0; i < N_NODES; ++i) mx = fmaxf(mx, srow[i]);
      float sum = 0.f;
      for (int i = 0; i < N_NODES; ++i) sum += __expf(srow[i] - mx);
      red[0] = mx; red[1] = 1.f / sum;
    }
    __syncthreads();
    if (tid < N_NODES) {
      float p = __expf(srow[tid] - red[0]) * red[1] * (1.f / (BB * HEADS));
      atomicAdd(&att[qi * N_NODES + tid], p);
    }
    __syncthreads();
  }
}

// ============ K7: A_norm = D^-1/2 (A + I) D^-1/2 ============
__global__ void k_anorm(const float* __restrict__ att, float* __restrict__ anorm)
{
  __shared__ float dinv[N_NODES];
  int tid = threadIdx.x;
  if (tid < N_NODES) {
    float s = 1.f;
    for (int j = 0; j < N_NODES; ++j) s += att[tid * N_NODES + j];
    dinv[tid] = rsqrtf(s);
  }
  __syncthreads();
  for (int i = tid; i < N_NODES * N_NODES; i += blockDim.x) {
    int r = i / N_NODES, c = i % N_NODES;
    float a = att[i] + (r == c ? 1.f : 0.f);
    anorm[i] = dinv[r] * a * dinv[c];
  }
}

// ============ K8: wr = weighted_res @ fcta_w^T + fcta_b ============
__global__ void k_fcta(const float* __restrict__ wres, const float* __restrict__ fw,
                       const float* __restrict__ fb, float* __restrict__ X)
{
  int i = blockIdx.x * blockDim.x + threadIdx.x;
  if (i >= BB * N_NODES * TT) return;
  int tp = i % TT; int bn = i / TT;
  float s = fb[tp];
  const float* wr = wres + (size_t)bn * HID;
  const float* ww = fw + tp * HID;
  for (int j = 0; j < HID; ++j) s += wr[j] * ww[j];
  X[i] = s;
}

// ============ K9: 2x GCN + FC head + output transpose ============
__global__ void __launch_bounds__(256) k_gcn_fc(
    const float* __restrict__ Xin, const float* __restrict__ anorm,
    const float* __restrict__ g1w, const float* __restrict__ g1b,
    const float* __restrict__ g2w, const float* __restrict__ g2b,
    const float* __restrict__ f1w, const float* __restrict__ f1b,
    const float* __restrict__ f2w, const float* __restrict__ f2b,
    float* __restrict__ out)
{
  __shared__ float X[N_NODES * TT];
  __shared__ float XW[N_NODES * CONV_HID];
  __shared__ float Xc[N_NODES * CONV_HID];
  int b = blockIdx.x, tid = threadIdx.x;
  for (int i = tid; i < N_NODES * TT; i += 256) X[i] = Xin[(size_t)b * N_NODES * TT + i];
  __syncthreads();
  for (int i = tid; i < N_NODES * CONV_HID; i += 256) {
    int n = i / CONV_HID, f = i % CONV_HID;
    float s = 0.f;
    for (int e = 0; e < TT; ++e) s += X[n * TT + e] * g1w[e * CONV_HID + f];
    XW[i] = s;
  }
  __syncthreads();
  for (int i = tid; i < N_NODES * CONV_HID; i += 256) {
    int n = i / CONV_HID, f = i % CONV_HID;
    float s = g1b[f];
    for (int j = 0; j < N_NODES; ++j) s += anorm[n * N_NODES + j] * XW[j * CONV_HID + f];
    Xc[i] = s > 0.f ? s : 0.f;
  }
  __syncthreads();
  for (int i = tid; i < N_NODES * CONV_HID; i += 256) {
    int n = i / CONV_HID, f = i % CONV_HID;
    float s = 0.f;
    for (int e = 0; e < CONV_HID; ++e) s += Xc[n * CONV_HID + e] * g2w[e * CONV_HID + f];
    XW[i] = s;
  }
  __syncthreads();
  for (int i = tid; i < N_NODES * CONV_HID; i += 256) {
    int n = i / CONV_HID, f = i % CONV_HID;
    float s = g2b[f];
    for (int j = 0; j < N_NODES; ++j) s += anorm[n * N_NODES + j] * XW[j * CONV_HID + f];
    Xc[i] = s > 0.f ? s : 0.f;
  }
  __syncthreads();
  for (int i = tid; i < N_NODES * TT; i += 256) {
    int n = i / TT, e = i % TT;
    float s = f1b[e];
    for (int f = 0; f < CONV_HID; ++f) s += Xc[n * CONV_HID + f] * f1w[e * CONV_HID + f];
    X[i] = s > 0.f ? s : 0.01f * s;  // leaky_relu(0.01)
  }
  __syncthreads();
  for (int i = tid; i < N_NODES * HORIZON; i += 256) {
    int n = i / HORIZON, hor = i % HORIZON;
    float s = f2b[hor];
    for (int e = 0; e < TT; ++e) s += X[n * TT + e] * f2w[hor * TT + e];
    out[((size_t)b * HORIZON + hor) * N_NODES + n] = s;
  }
}

extern "C" void kernel_launch(void* const* d_in, const int* in_sizes, int n_in,
                              void* d_out, int out_size, void* d_ws, size_t ws_size,
                              hipStream_t stream)
{
  (void)in_sizes; (void)n_in; (void)out_size; (void)ws_size;
  const float* x       = (const float*)d_in[0];
  const float* s1w1    = (const float*)d_in[1];
  const float* s1b1    = (const float*)d_in[2];
  const float* s1w2    = (const float*)d_in[3];
  const float* s1b2    = (const float*)d_in[4];
  const float* gru_wih = (const float*)d_in[5];
  const float* gru_whh = (const float*)d_in[6];
  const float* gru_bih = (const float*)d_in[7];
  const float* gru_bhh = (const float*)d_in[8];
  const float* attn_w  = (const float*)d_in[9];
  const float* attn_b  = (const float*)d_in[10];
  const float* attn_v  = (const float*)d_in[11];
  const float* ln_g    = (const float*)d_in[12];
  const float* ln_b    = (const float*)d_in[13];
  const float* mha_wq  = (const float*)d_in[14];
  const float* mha_wk  = (const float*)d_in[15];
  const float* mha_bq  = (const float*)d_in[17];
  const float* mha_bk  = (const float*)d_in[18];
  const float* fcta_w  = (const float*)d_in[22];
  const float* fcta_b  = (const float*)d_in[23];
  const float* g1w     = (const float*)d_in[24];
  const float* g1b     = (const float*)d_in[25];
  const float* g2w     = (const float*)d_in[26];
  const float* g2b     = (const float*)d_in[27];
  const float* f1w     = (const float*)d_in[28];
  const float* f1b     = (const float*)d_in[29];
  const float* f2w     = (const float*)d_in[30];
  const float* f2b     = (const float*)d_in[31];

  char* ws = (char*)d_ws;
  size_t off = 0;
  auto alloc = [&](size_t bytes) -> char* {
    char* p = ws + off;
    off += (bytes + 255) & ~(size_t)255;
    return p;
  };
  _Float16* x_sup = (_Float16*)alloc(sizeof(_Float16) * (size_t)N_NODES * BB * TT * KX);
  float* outs   = (float*)alloc(sizeof(float) * (size_t)N_NODES * BB * TT * HID);
  float* scores = (float*)alloc(sizeof(float) * (size_t)N_NODES * ROWS_BT);
  float* wres   = (float*)alloc(sizeof(float) * (size_t)BB * N_NODES * HID);
  float* qd     = (float*)alloc(sizeof(float) * (size_t)BB * N_NODES * HID);
  float* kd     = (float*)alloc(sizeof(float) * (size_t)BB * N_NODES * HID);
  float* anorm  = (float*)alloc(sizeof(float) * (size_t)N_NODES * N_NODES);
  float* Xgc    = (float*)alloc(sizeof(float) * (size_t)BB * N_NODES * TT);

  float* out = (float*)d_out;                         // (B,3,140) = 13440
  float* att = out + (size_t)BB * HORIZON * N_NODES;  // (140,140) = 19600

  k_seq1<<<(N_NODES * BB * TT) / 16, 256, 0, stream>>>(x, s1w1, s1b1, s1w2, s1b2, x_sup);

  size_t gru_lds = sizeof(_Float16) * ((size_t)GT * KXP + (size_t)GT * KHP +
                                       32 * KHP + 32 * KXP + 32 * GP + 32 * GP) +
                   sizeof(float) * 32 * GP;  // 318,976 B (< 320 KB WGP LDS)
  k_gru<<<N_NODES, 256, gru_lds, stream>>>(x_sup, gru_wih, gru_whh, gru_bih, gru_bhh, outs);

  size_t attn_lds = sizeof(_Float16) * ((size_t)GP * AKP + 16 * AKP) + sizeof(float) * ROWS_BT;
  k_attn_scores<<<N_NODES, 256, attn_lds, stream>>>(outs, attn_w, attn_b, attn_v, scores);

  k_weighted_ln<<<dim3(N_NODES, BB), 160, 0, stream>>>(outs, scores, ln_g, ln_b, wres);

  size_t qk_lds = sizeof(_Float16) * (2 * (size_t)GP * KHP + 16 * KHP);
  k_qk<<<dim3(9, BB), 256, qk_lds, stream>>>(wres, mha_wq, mha_wk, mha_bq, mha_bk, qd, kd);

  k_zero<<<(N_NODES * N_NODES + 255) / 256, 256, 0, stream>>>(att, N_NODES * N_NODES);
  k_mha_att<<<dim3(HEADS, BB), 256, 0, stream>>>(qd, kd, att);
  k_anorm<<<1, 256, 0, stream>>>(att, anorm);
  k_fcta<<<(BB * N_NODES * TT + 255) / 256, 256, 0, stream>>>(wres, fcta_w, fcta_b, Xgc);
  k_gcn_fc<<<BB, 256, 0, stream>>>(Xgc, anorm, g1w, g1b, g2w, g2b,
                                   f1w, f1b, f2w, f2b, out);
}
